// CompetitionModel_56719338111750
// MI455X (gfx1250) — matrix operand order repaired and checked by reading, not compiled, not run on hardware
//
#include <hip/hip_runtime.h>

// ---------------------------------------------------------------------------
// SNN forward: 2 populations, E=1024 LIF neurons each, B=256, D=512, 567 steps
// Recurrent + input GEMMs via v_wmma_f32_16x16x32_bf16 (spikes are exact in
// bf16; weights converted once).  Weights (6MB bf16) stay L2-resident.
// ---------------------------------------------------------------------------

#define BB 256
#define TT 512
#define DD 512
#define EE 1024
#define TP 567          // TT + EXTRA(5) + NB_DEC(50)
#define NBDEC 50

typedef __attribute__((ext_vector_type(16))) __bf16 v16bf;
typedef __attribute__((ext_vector_type(8)))  float  v8f;

__device__ __forceinline__ unsigned short f32_to_bf16(float f) {
    unsigned u = __builtin_bit_cast(unsigned, f);
    unsigned r = 0x7FFFu + ((u >> 16) & 1u);
    return (unsigned short)((u + r) >> 16);
}

// A-matrix tile (16 rows x 32 K, bf16, row-major source, stride in elements).
// ISA layout: lane<16 -> row=lane,   K = {k0..k0+7, k0+16..k0+23}
//             lane>=16 -> row=lane-16, K = {k0+8..k0+15, k0+24..k0+31}
__device__ __forceinline__ v16bf load_a_tile(const __bf16* __restrict__ base,
                                             int stride, int row0, int k0, int lane) {
    int r  = row0 + (lane & 15);
    int kb = k0 + ((lane >> 4) << 3);          // +0 or +8
    const __bf16* p = base + (size_t)r * stride + kb;
    union { v16bf v; uint4 q[2]; } u;
    u.q[0] = *(const uint4*)(p);
    u.q[1] = *(const uint4*)(p + 16);
    return u.v;
}

// B-matrix tile (32 K x 16 cols).  B[k][n] = W[j0+n][k] for W row-major [E,K].
// ISA layout: lane holds col n=lane&15, K = {kb..kb+7, kb+16..kb+23}
//             with kb = k0 (+8 if lane>=16).
__device__ __forceinline__ v16bf load_b_tile(const __bf16* __restrict__ W,
                                             int stride, int j0, int k0, int lane) {
    int n  = j0 + (lane & 15);
    int kb = k0 + ((lane >> 4) << 3);          // +0 or +8
    const __bf16* p = W + (size_t)n * stride + kb;
    union { v16bf v; uint4 q[2]; } u;
    u.q[0] = *(const uint4*)(p);
    u.q[1] = *(const uint4*)(p + 16);
    return u.v;
}

// ---------------------------------------------------------------------------
// Init: convert weights + x[:,0,:] to bf16, zero all state.
// ---------------------------------------------------------------------------
__global__ void snn_init(const float* __restrict__ x,
                         const float* __restrict__ Win1, const float* __restrict__ Wrec1,
                         const float* __restrict__ Win2, const float* __restrict__ Wrec2,
                         unsigned short* win1b, unsigned short* wrec1b,
                         unsigned short* win2b, unsigned short* wrec2b,
                         unsigned short* xt_bf, unsigned short* spk, float* mem,
                         float* p_ro, float* p_inh, float* ro,
                         float* mem_inh, float* inh_spk, float* out_acc) {
    size_t tid = (size_t)blockIdx.x * blockDim.x + threadIdx.x;
    size_t str = (size_t)gridDim.x * blockDim.x;
    for (size_t i = tid; i < (size_t)EE * DD; i += str) {
        win1b[i] = f32_to_bf16(Win1[i]);
        win2b[i] = f32_to_bf16(Win2[i]);
    }
    for (size_t i = tid; i < (size_t)EE * EE; i += str) {
        wrec1b[i] = f32_to_bf16(Wrec1[i]);
        wrec2b[i] = f32_to_bf16(Wrec2[i]);
    }
    for (size_t i = tid; i < (size_t)BB * DD; i += str) {   // t = 0 slice
        size_t b = i >> 9, d = i & 511;
        xt_bf[i] = f32_to_bf16(x[(b * TT) * DD + d]);
    }
    for (size_t i = tid; i < 4ull * BB * EE; i += str) spk[i] = 0;       // 2 bufs x 2 pops
    for (size_t i = tid; i < 2ull * BB * EE; i += str) mem[i] = 0.0f;
    for (size_t i = tid; i < 2ull * BB * 64; i += str) { p_ro[i] = 0.0f; p_inh[i] = 0.0f; }
    for (size_t i = tid; i < 2ull * BB; i += str) { ro[i] = 0.0f; out_acc[i] = 0.0f; }
    for (size_t i = tid; i < (size_t)BB; i += str) { mem_inh[i] = 0.0f; inh_spk[i] = 0.0f; }
}

// ---------------------------------------------------------------------------
// Kernel A: one wave per 16x16 output tile.  2048 tiles = 2 pops x 16 x 64.
// accI = xt @ Win^T  (K=512), accR = spk_prev @ Wrec^T (K=1024), separate
// accumulators because abs() is applied per term.  Fused LIF + spike write +
// deterministic per-(row, j-tile) partial sums for readout / inhibitory dots.
// ---------------------------------------------------------------------------
__global__ __launch_bounds__(256) void snn_step_gemm(
    const __bf16* __restrict__ xt,
    const __bf16* __restrict__ win1, const __bf16* __restrict__ wrec1,
    const __bf16* __restrict__ win2, const __bf16* __restrict__ wrec2,
    const __bf16* __restrict__ spk_rd, unsigned short* __restrict__ spk_wr,
    float* __restrict__ mem, const float* __restrict__ inh_spk,
    const float* __restrict__ wfb1, const float* __restrict__ wfb2,
    const float* __restrict__ wro1, const float* __restrict__ wro2,
    const float* __restrict__ winh,
    float* __restrict__ p_ro, float* __restrict__ p_inh) {

    const int lane = threadIdx.x & 31;
    const int wave = threadIdx.x >> 5;
    const int tile = blockIdx.x * 8 + wave;
    const int pop  = tile >> 10;
    const int rem  = tile & 1023;
    const int it   = rem >> 6;        // 0..15  (B tiles)
    const int jt   = rem & 63;        // 0..63  (E tiles)
    const int i0   = it << 4, j0 = jt << 4;

    const __bf16* win   = pop ? win2  : win1;
    const __bf16* wrec  = pop ? wrec2 : wrec1;
    const float*  wfb   = pop ? wfb2  : wfb1;
    const float*  wro   = pop ? wro2  : wro1;
    const __bf16* spin  = spk_rd + (size_t)pop * (BB * EE);
    unsigned short* spo = spk_wr + (size_t)pop * (BB * EE);
    float* memp = mem  + (size_t)pop * (BB * EE);
    float* prop = p_ro + (size_t)pop * (BB * 64);
    float* pinp = p_inh + (size_t)pop * (BB * 64);

    v8f accI = {};
    for (int k0 = 0; k0 < DD; k0 += 32) {
        v16bf a = load_a_tile(xt, DD, i0, k0, lane);
        v16bf b = load_b_tile(win, DD, j0, k0, lane);
        accI = __builtin_amdgcn_wmma_f32_16x16x32_bf16(false, a, false, b,
                                                       (short)0, accI, false, false);
    }
    v8f accR = {};
    for (int k0 = 0; k0 < EE; k0 += 32) {
        v16bf a = load_a_tile(spin, EE, i0, k0, lane);
        v16bf b = load_b_tile(wrec, EE, j0, k0, lane);
        accR = __builtin_amdgcn_wmma_f32_16x16x32_bf16(false, a, false, b,
                                                       (short)0, accR, false, false);
    }

    // Epilogue.  C/D layout: VGPR v -> row i0+v (+8 if lane>=16), col j0+(lane&15).
    const int nloc = lane & 15;
    const int hi   = lane >> 4;
    const int n    = j0 + nloc;
    const float afb   = fabsf(wfb[n]);   // |W_inhfb[e]| ; inh_spk in {0,1}
    const float wron  = wro[n];
    const float winhn = winh[n];

    #pragma unroll
    for (int v = 0; v < 8; ++v) {
        const int m = i0 + v + (hi << 3);
        const size_t idx = (size_t)m * EE + n;
        const float curr = fabsf(accI[v]) + fabsf(accR[v]) - inh_spk[m] * afb;
        const float mo = memp[idx];
        const float reset = mo > 1.0f ? 1.0f : 0.0f;            // THR = 1
        const float mn = 0.9f * mo + curr - reset;              // BETA_EXT = 0.9
        memp[idx] = mn;
        const float s = (mn - 1.0f) > 0.0f ? 1.0f : 0.0f;
        spo[idx] = s != 0.0f ? (unsigned short)0x3F80 : (unsigned short)0;

        float cro = s * wron;
        float cin = s * winhn;
        #pragma unroll
        for (int o = 8; o > 0; o >>= 1) {                       // reduce 16 N-lanes
            cro += __shfl_xor(cro, o, 16);
            cin += __shfl_xor(cin, o, 16);
        }
        if (nloc == 0) {                                        // lanes 0 and 16
            prop[(size_t)m * 64 + jt] = cro;
            pinp[(size_t)m * 64 + jt] = cin;
        }
    }
}

// ---------------------------------------------------------------------------
// Kernel B: block 0 -> per-batch readout & inhibitory LIF update + output
// accumulation; blocks 1..64 -> convert x[:, t+1, :] to bf16 (zeros past TT).
// ---------------------------------------------------------------------------
__global__ __launch_bounds__(256) void snn_step_scalar(
    int t, const float* __restrict__ x, unsigned short* __restrict__ xt_bf,
    const float* __restrict__ p_ro, const float* __restrict__ p_inh,
    float* __restrict__ ro, float* __restrict__ mem_inh,
    float* __restrict__ inh_spk, float* __restrict__ out_acc) {

    if (blockIdx.x == 0) {
        const int b = threadIdx.x;
        float s1 = 0.f, s2 = 0.f, h1 = 0.f, h2 = 0.f;
        #pragma unroll 8
        for (int j = 0; j < 64; ++j) {
            s1 += p_ro[(size_t)b * 64 + j];
            s2 += p_ro[(size_t)(BB * 64) + (size_t)b * 64 + j];
            h1 += p_inh[(size_t)b * 64 + j];
            h2 += p_inh[(size_t)(BB * 64) + (size_t)b * 64 + j];
        }
        const float r1 = 0.95f * ro[b]      + fabsf(s1);        // BETA_RO
        const float r2 = 0.95f * ro[BB + b] + fabsf(s2);
        ro[b] = r1; ro[BB + b] = r2;
        const float ci = fabsf(h1) + fabsf(h2);
        const float mo = mem_inh[b];
        const float reset = mo > 1.0f ? 1.0f : 0.0f;
        const float mn = 0.9f * mo + ci - reset;                // BETA_INH = 0.9
        mem_inh[b] = mn;
        inh_spk[b] = (mn - 1.0f) > 0.0f ? 1.0f : 0.0f;
        if (t >= TP - NBDEC) {
            out_acc[b * 2 + 0] += r1;
            out_acc[b * 2 + 1] += r2;
        }
    } else {
        const int tn = t + 1;
        const int tid = (blockIdx.x - 1) * 256 + threadIdx.x;   // 16384 threads
        #pragma unroll
        for (int q = 0; q < 8; ++q) {
            const int e = tid + q * 16384;                      // e < 131072
            const int b = e >> 9, d = e & 511;
            const float v = (tn < TT) ? x[((size_t)b * TT + tn) * DD + d] : 0.0f;
            xt_bf[e] = f32_to_bf16(v);
        }
    }
}

__global__ void snn_final(const float* __restrict__ acc, float* __restrict__ out) {
    int i = blockIdx.x * blockDim.x + threadIdx.x;
    if (i < BB * 2) out[i] = acc[i] * (1.0f / (float)NBDEC);
}

// ---------------------------------------------------------------------------
extern "C" void kernel_launch(void* const* d_in, const int* in_sizes, int n_in,
                              void* d_out, int out_size, void* d_ws, size_t ws_size,
                              hipStream_t stream) {
    const float* x     = (const float*)d_in[0];
    const float* Win1  = (const float*)d_in[1];
    const float* Wrec1 = (const float*)d_in[2];
    const float* Wfb1  = (const float*)d_in[3];
    const float* Wro1  = (const float*)d_in[4];
    const float* Win2  = (const float*)d_in[5];
    const float* Wrec2 = (const float*)d_in[6];
    const float* Wfb2  = (const float*)d_in[7];
    const float* Wro2  = (const float*)d_in[8];
    const float* Winh  = (const float*)d_in[9];

    char* ws = (char*)d_ws;
    size_t off = 0;
    auto alloc = [&](size_t bytes) { size_t r = (off + 255) & ~(size_t)255; off = r + bytes; return r; };

    unsigned short* win1b = (unsigned short*)(ws + alloc((size_t)EE * DD * 2));
    unsigned short* wrec1b = (unsigned short*)(ws + alloc((size_t)EE * EE * 2));
    unsigned short* win2b = (unsigned short*)(ws + alloc((size_t)EE * DD * 2));
    unsigned short* wrec2b = (unsigned short*)(ws + alloc((size_t)EE * EE * 2));
    unsigned short* xt_bf = (unsigned short*)(ws + alloc((size_t)BB * DD * 2));
    unsigned short* spk   = (unsigned short*)(ws + alloc(4ull * BB * EE * 2)); // [buf][pop]
    float* mem     = (float*)(ws + alloc(2ull * BB * EE * 4));                 // [pop]
    float* p_ro    = (float*)(ws + alloc(2ull * BB * 64 * 4));
    float* p_inh   = (float*)(ws + alloc(2ull * BB * 64 * 4));
    float* ro      = (float*)(ws + alloc(2ull * BB * 4));
    float* mem_inh = (float*)(ws + alloc((size_t)BB * 4));
    float* inh_spk = (float*)(ws + alloc((size_t)BB * 4));
    float* out_acc = (float*)(ws + alloc(2ull * BB * 4));

    snn_init<<<512, 256, 0, stream>>>(x, Win1, Wrec1, Win2, Wrec2,
                                      win1b, wrec1b, win2b, wrec2b,
                                      xt_bf, spk, mem, p_ro, p_inh, ro,
                                      mem_inh, inh_spk, out_acc);

    for (int t = 0; t < TP; ++t) {
        const __bf16* spk_rd = (const __bf16*)spk + (size_t)(t & 1) * (2ull * BB * EE);
        unsigned short* spk_wr = spk + (size_t)((t + 1) & 1) * (2ull * BB * EE);
        snn_step_gemm<<<256, 256, 0, stream>>>(
            (const __bf16*)xt_bf,
            (const __bf16*)win1b, (const __bf16*)wrec1b,
            (const __bf16*)win2b, (const __bf16*)wrec2b,
            spk_rd, spk_wr, mem, inh_spk,
            Wfb1, Wfb2, Wro1, Wro2, Winh, p_ro, p_inh);
        snn_step_scalar<<<65, 256, 0, stream>>>(
            t, x, xt_bf, p_ro, p_inh, ro, mem_inh, inh_spk, out_acc);
    }

    snn_final<<<2, 256, 0, stream>>>(out_acc, (float*)d_out);
}
